// Model_49658411876880
// MI455X (gfx1250) — matrix operand order repaired and checked by reading, not compiled, not run on hardware
//
#include <hip/hip_runtime.h>

// ---------------------------------------------------------------------------
// Problem constants (from reference)
// ---------------------------------------------------------------------------
#define HH 128
#define WW 128
#define HW 16384            // H*W
#define VV 16               // object types
#define AA 8                // action types
#define PP 64               // programs
#define CC 3                // input channels
#define MAXT 50

typedef __attribute__((ext_vector_type(16))) _Float16 v16h;
typedef __attribute__((ext_vector_type(8)))  float    v8f;

struct KeySet { unsigned k[12]; };   // 6 threefry subkeys (k0,k1 pairs)

// ---------------------------------------------------------------------------
// Threefry2x32 (JAX-compatible core), usable host+device
// ---------------------------------------------------------------------------
__host__ __device__ inline unsigned rotl32(unsigned v, int r) {
    return (v << r) | (v >> (32 - r));
}

__host__ __device__ inline void threefry2x32(unsigned k0, unsigned k1,
                                             unsigned x0, unsigned x1,
                                             unsigned& o0, unsigned& o1) {
    unsigned ks2 = k0 ^ k1 ^ 0x1BD11BDAu;
    x0 += k0; x1 += k1;
#define TF_RND(r) { x0 += x1; x1 = rotl32(x1, r); x1 ^= x0; }
    TF_RND(13) TF_RND(15) TF_RND(26) TF_RND(6)   x0 += k1;  x1 += ks2 + 1u;
    TF_RND(17) TF_RND(29) TF_RND(16) TF_RND(24)  x0 += ks2; x1 += k0  + 2u;
    TF_RND(13) TF_RND(15) TF_RND(26) TF_RND(6)   x0 += k0;  x1 += k1  + 3u;
    TF_RND(17) TF_RND(29) TF_RND(16) TF_RND(24)  x0 += k1;  x1 += ks2 + 4u;
    TF_RND(13) TF_RND(15) TF_RND(26) TF_RND(6)   x0 += ks2; x1 += k0  + 5u;
#undef TF_RND
    o0 = x0; o1 = x1;
}

// JAX random_bits: flat array of size S, counter iota split into halves.
// word[i] for i<S/2 is lane-0 output of threefry(key,(i,i+S/2)); else lane-1.
__device__ inline unsigned random_word(unsigned k0, unsigned k1,
                                       unsigned idx, unsigned half) {
    unsigned o0, o1;
    if (idx < half) { threefry2x32(k0, k1, idx, idx + half, o0, o1); return o0; }
    threefry2x32(k0, k1, idx - half, idx, o0, o1); return o1;
}

// Gumbel-argmax categorical draw, matching jax.random.categorical semantics.
__device__ inline int categorical_draw(unsigned k0, unsigned k1, unsigned n,
                                       int V, const float* __restrict__ logits,
                                       unsigned S) {
    unsigned half = S >> 1;
    float best = -3.4e38f;
    int bi = 0;
#pragma unroll 4
    for (int v = 0; v < V; ++v) {
        unsigned bits = random_word(k0, k1, n * (unsigned)V + (unsigned)v, half);
        float u = __uint_as_float((bits >> 9) | 0x3f800000u) - 1.0f;   // [0,1)
        u = u * (1.0f - 1.17549435e-38f) + 1.17549435e-38f;            // (tiny,1)
        float g = -__logf(-__logf(u));
        float s = logits[v] + g;
        if (s > best) { best = s; bi = v; }   // first-max wins (jnp.argmax)
    }
    return bi;
}

// ---------------------------------------------------------------------------
// Kernel 1: conv3x3 (implicit GEMM via WMMA f16->f32) + bias + channel argmax
// One wave = 16 consecutive pixels of one row; M=16, K=27 (pad 32), N=16.
// ---------------------------------------------------------------------------
__global__ __launch_bounds__(128)
void conv_argmax_wmma(const float* __restrict__ obs,    // [3,128,128]
                      const float* __restrict__ w,      // [16,3,3,3]
                      const float* __restrict__ bias,   // [16]
                      unsigned char* __restrict__ grid) // [128*128]
{
    __shared__ float sc[4][16 * 16];
    const int lane = threadIdx.x & 31;
    const int wv   = threadIdx.x >> 5;
    const int tile = blockIdx.x * 4 + wv;     // 0..1023
    const int px0  = tile << 4;               // base pixel (row-major)
    const int y    = px0 >> 7;
    const int x0   = px0 & 127;

    // ---- A fragment: 16x32 f16, lanes 0-15 hold K-block {0-7,16-23},
    //      lanes 16-31 hold K-block {8-15,24-31}; row M = lane&15.
    const int m  = lane & 15;
    const int kb = (lane >> 4) ? 8 : 0;
    v16h a;
#pragma unroll
    for (int h = 0; h < 16; ++h) {
        int vg = h >> 1, sub = h & 1;
        int K = (vg < 4) ? (kb + 2 * vg + sub) : (16 + kb + 2 * (vg - 4) + sub);
        float val = 0.0f;
        if (K < 27) {
            int c = K / 9, r9 = K % 9, ky = r9 / 3, kx = r9 % 3;
            int yy = y + ky - 1, xx = x0 + m + kx - 1;
            if ((unsigned)yy < 128u && (unsigned)xx < 128u)
                val = obs[(c << 14) + (yy << 7) + xx];
        }
        a[h] = (_Float16)val;
    }

    // ---- B fragment: 32x16 f16, column n = lane&15, lanes 0-15: K {0-7,16-23},
    //      lanes 16-31: K {8-15,24-31}.  B[K][n] = conv_w[n][K].
    const int n   = lane & 15;
    const int kb2 = (lane >> 4) ? 8 : 0;
    v16h bm;
#pragma unroll
    for (int h = 0; h < 16; ++h) {
        int K = (h < 8) ? (kb2 + h) : (16 + kb2 + h - 8);
        bm[h] = (_Float16)((K < 27) ? w[n * 27 + K] : 0.0f);
    }

    v8f acc = {};
    acc = __builtin_amdgcn_wmma_f32_16x16x32_f16(false, a, false, bm,
                                                 (short)0, acc, false, false);

    // ---- D layout: lane gives N (=lane&15); VGPR i gives M (=i or i+8).
    const int Nc = lane & 15;
    const int Mb = (lane >> 4) * 8;
    const float bN = bias[Nc];
#pragma unroll
    for (int i = 0; i < 8; ++i)
        sc[wv][(Mb + i) * 16 + Nc] = acc[i] + bN;
    __syncthreads();

    if (lane < 16) {
        const float* row = &sc[wv][lane * 16];
        float best = row[0];
        int bi = 0;
#pragma unroll
        for (int ch = 1; ch < 16; ++ch)
            if (row[ch] > best) { best = row[ch]; bi = ch; }
        grid[px0 + lane] = (unsigned char)bi;
    }
}

// ---------------------------------------------------------------------------
// Kernel 2a/2b: init workspace; first row-major occurrence per object value.
// ---------------------------------------------------------------------------
__global__ void init_ws(int* __restrict__ first, float* __restrict__ counts) {
    int t = threadIdx.x;
    if (t < 16) first[t] = 0x7fffffff;
    if (t < 8)  counts[t] = 0.0f;
}

__global__ __launch_bounds__(256)
void first_occurrence(const unsigned char* __restrict__ grid,
                      int* __restrict__ first) {
    int i = blockIdx.x * 256 + threadIdx.x;
    if (i < HW) atomicMin(&first[grid[i]], i);
}

// ---------------------------------------------------------------------------
// Kernel 3: 1M rays. Grid + all program tables staged in LDS (CDNA5: 320KB/WGP).
// ---------------------------------------------------------------------------
__global__ __launch_bounds__(256)
void ray_march(const unsigned char* __restrict__ gridg,
               const int* __restrict__ firstg,
               const float* __restrict__ w_obj, const float* __restrict__ w_pos,
               const float* __restrict__ w_neg, const float* __restrict__ w_act,
               const float* __restrict__ w_dir, const float* __restrict__ w_not,
               float* __restrict__ counts, KeySet ks)
{
    __shared__ float         tb[4224];     // obj|pos|neg|act|dir|not tables
    __shared__ unsigned char g[HW];        // 16KB object grid
    __shared__ int           sfirst[16];
    __shared__ float         lc[8];

    const int tid = threadIdx.x;
    __builtin_prefetch(gridg, 0, 0);       // global_prefetch_b8

    for (int i = tid; i < 1024; i += 256) {
        tb[i]        = w_obj[i];
        tb[1024 + i] = w_pos[i];
        tb[2048 + i] = w_neg[i];
    }
    for (int i = tid; i < 512; i += 256) {
        tb[3072 + i] = w_act[i];
        tb[3584 + i] = w_dir[i];
    }
    if (tid < 128) tb[4096 + tid] = w_not[tid];
    {
        const uint4* gs = (const uint4*)gridg;
        uint4*       gd = (uint4*)g;
        for (int i = tid; i < HW / 16; i += 256) gd[i] = gs[i];
    }
    if (tid < 16) sfirst[tid] = firstg[tid];
    if (tid < 8)  lc[tid] = 0.0f;
    __syncthreads();

    const unsigned nidx = blockIdx.x * 256u + (unsigned)tid;  // 0..HW*P-1
    const int p = (int)(nidx & 63u);

    const int s_obj = categorical_draw(ks.k[0],  ks.k[1],  nidx, 16, &tb[p * 16],         16777216u);
    const int s_pos = categorical_draw(ks.k[2],  ks.k[3],  nidx, 16, &tb[1024 + p * 16],  16777216u);
    const int s_neg = categorical_draw(ks.k[4],  ks.k[5],  nidx, 16, &tb[2048 + p * 16],  16777216u);
    const int s_act = categorical_draw(ks.k[6],  ks.k[7],  nidx, 8,  &tb[3072 + p * 8],   8388608u);
    const int s_dir = categorical_draw(ks.k[8],  ks.k[9],  nidx, 8,  &tb[3584 + p * 8],   8388608u);
    const int s_not = categorical_draw(ks.k[10], ks.k[11], nidx, 2,  &tb[4096 + p * 2],   2097152u);

    bool cond = false;
    const int f = sfirst[s_obj];
    if (f != 0x7fffffff) {                 // object exists
        int r = f >> 7, c = f & 127;
        const int drs[8] = { 1, 0, -1, 0, 1, -1, 1, -1 };
        const int dcs[8] = { 0, 1, 0, -1, 1,  1, -1, -1 };
        const int dr = drs[s_dir], dc = dcs[s_dir];
        for (int t = 0; t < MAXT; ++t) {
            r += dr; c += dc;
            if ((unsigned)r >= 128u || (unsigned)c >= 128u) break;  // ~inb stop
            const int val = g[(r << 7) + c];
            if (val == s_pos) { cond = true; break; }               // hit_pos
            if (val == s_neg) break;                                // hit_neg
        }
    }
    cond ^= (s_not == 1);

    atomicAdd(&lc[s_act], cond ? 1.0f : 0.0f);
    __syncthreads();
    if (tid < 8) atomicAdd(&counts[tid], lc[tid]);
}

// ---------------------------------------------------------------------------
// Kernel 4: softmax over 8 action counts
// ---------------------------------------------------------------------------
__global__ void softmax8(const float* __restrict__ counts,
                         float* __restrict__ out) {
    if (threadIdx.x == 0) {
        float m = counts[0];
        for (int i = 1; i < 8; ++i) m = fmaxf(m, counts[i]);
        float e[8], s = 0.0f;
        for (int i = 0; i < 8; ++i) { e[i] = __expf(counts[i] - m); s += e[i]; }
        float inv = 1.0f / s;
        for (int i = 0; i < 8; ++i) out[i] = e[i] * inv;
    }
}

// ---------------------------------------------------------------------------
// Launch
// ---------------------------------------------------------------------------
extern "C" void kernel_launch(void* const* d_in, const int* in_sizes, int n_in,
                              void* d_out, int out_size, void* d_ws, size_t ws_size,
                              hipStream_t stream) {
    const float* obs    = (const float*)d_in[0];
    const float* conv_w = (const float*)d_in[1];
    const float* conv_b = (const float*)d_in[2];
    const float* w_obj  = (const float*)d_in[3];
    const float* w_pos  = (const float*)d_in[4];
    const float* w_neg  = (const float*)d_in[5];
    const float* w_act  = (const float*)d_in[6];
    const float* w_dir  = (const float*)d_in[7];
    const float* w_not  = (const float*)d_in[8];

    unsigned char* grid = (unsigned char*)d_ws;
    int*   first  = (int*)((char*)d_ws + HW);
    float* counts = (float*)((char*)d_ws + HW + 64);
    float* out    = (float*)d_out;

    // jax.random.key(42) -> (0,42); split into 6 subkeys (threefry counter mode,
    // iota(12) split into halves, output halves concatenated then paired).
    unsigned h0[6], h1[6], flat[12];
    for (int i = 0; i < 6; ++i)
        threefry2x32(0u, 42u, (unsigned)i, (unsigned)(i + 6), h0[i], h1[i]);
    for (int i = 0; i < 6; ++i) { flat[i] = h0[i]; flat[6 + i] = h1[i]; }
    KeySet ks;
    for (int j = 0; j < 12; ++j) ks.k[j] = flat[j];

    conv_argmax_wmma<<<256, 128, 0, stream>>>(obs, conv_w, conv_b, grid);
    init_ws<<<1, 32, 0, stream>>>(first, counts);
    first_occurrence<<<HW / 256, 256, 0, stream>>>(grid, first);
    ray_march<<<(HW * PP) / 256, 256, 0, stream>>>(grid, first, w_obj, w_pos,
                                                   w_neg, w_act, w_dir, w_not,
                                                   counts, ks);
    softmax8<<<1, 32, 0, stream>>>(counts, out);
}
